// AICASage_49735721288419
// MI455X (gfx1250) — matrix operand order repaired and checked by reading, not compiled, not run on hardware
//
#include <hip/hip_runtime.h>
#include <hip/hip_bf16.h>
#include <math.h>

// ---------------------------------------------------------------------------
// GraphSAGE 3-layer forward for MI455X (gfx1250, wave32, WMMA).
// Strategy: bf16 activations/weights (halves HBM traffic on the edge-scatter,
// which dominates at 23.3 TB/s), f32 atomic accumulation for the mean
// aggregation, and v_wmma_f32_16x16x32_bf16 for all GEMMs with fused
// bias + tanh + dropout epilogue.
// ---------------------------------------------------------------------------

typedef __attribute__((ext_vector_type(16))) __bf16 v16bf;
typedef __attribute__((ext_vector_type(8)))  __bf16 v8bf;
typedef __attribute__((ext_vector_type(8)))  float  v8f;

#define DIM 256          // IN_DIM == HID_DIM
#define OUTD3 128        // OUT_DIM
#define EPB 32           // edges per block in scatter

// ---------------------------- utility kernels ------------------------------

__global__ void k_zero_f32(float* __restrict__ p, size_t n) {
  size_t i = (size_t)blockIdx.x * blockDim.x + threadIdx.x;
  size_t stride = (size_t)gridDim.x * blockDim.x;
  for (; i < n; i += stride) p[i] = 0.0f;
}

__global__ void k_f32_to_bf16(const float* __restrict__ in,
                              __bf16* __restrict__ out, size_t n) {
  size_t i = (size_t)blockIdx.x * blockDim.x + threadIdx.x;
  size_t stride = (size_t)gridDim.x * blockDim.x;
  for (; i < n; i += stride) out[i] = (__bf16)in[i];
}

__global__ void k_deg(const int* __restrict__ dst, float* __restrict__ deg, int E) {
  int e = blockIdx.x * blockDim.x + threadIdx.x;
  if (e < E) atomicAdd(&deg[dst[e]], 1.0f);
}

__global__ void k_deg_inv(float* __restrict__ deg, int n) {
  int i = blockIdx.x * blockDim.x + threadIdx.x;
  if (i < n) deg[i] = 1.0f / fmaxf(deg[i], 1.0f);
}

// Mean-aggregate: sum[dst] += feat[src]  (f32 atomic accumulation).
// blockDim.x == DIM; thread t owns feature dim t for EPB consecutive edges.
__global__ void k_scatter(const __bf16* __restrict__ feat,
                          const int* __restrict__ src,
                          const int* __restrict__ dst,
                          float* __restrict__ sum, int E) {
  int t  = threadIdx.x;                 // feature dim
  int e0 = blockIdx.x * EPB;
  int ee = e0 + EPB; if (ee > E) ee = E;
  for (int e = e0; e < ee; ++e) {
    int s = src[e];
    int d = dst[e];
    if (e + 1 < ee) {                   // emits global_prefetch_b8 on gfx1250
      __builtin_prefetch(&feat[(size_t)src[e + 1] * DIM + t], 0, 1);
    }
    float v = (float)feat[(size_t)s * DIM + t];
    atomicAdd(&sum[(size_t)d * DIM + t], v);
  }
}

// mean_bf16[i] = bf16( sum[i] * deginv[i / DIM] )
__global__ void k_mean_bf16(const float* __restrict__ sum,
                            const float* __restrict__ deginv,
                            __bf16* __restrict__ meanb, size_t n) {
  size_t i = (size_t)blockIdx.x * blockDim.x + threadIdx.x;
  size_t stride = (size_t)gridDim.x * blockDim.x;
  for (; i < n; i += stride)
    meanb[i] = (__bf16)(sum[i] * deginv[i >> 8]);   // DIM == 256
}

// ------------------------------ WMMA GEMM ----------------------------------
// A-operand k order used here (16-bit, 16x32; contiguous k, not AMD's table order, same as B):
//   lane<16 : row M=lane,   K = k0..k0+15
//   lane>=16: row M=lane-16,K = k0+16..k0+31
__device__ __forceinline__ v16bf load_a16(const __bf16* p) {
  v8bf lo = *(const v8bf*)(p);          // 16B aligned
  v8bf hi = *(const v8bf*)(p + 8);
  v16bf r;
#pragma unroll
  for (int i = 0; i < 8; ++i) { r[i] = lo[i]; r[i + 8] = hi[i]; }
  return r;
}
// B-operand k order used here (16-bit, 32x16; contiguous k, same as A): lane = column N, 16 consecutive K.
__device__ __forceinline__ v16bf load_b16(const __bf16* p) {
  v8bf lo = *(const v8bf*)(p);
  v8bf hi = *(const v8bf*)(p + 8);
  v16bf r;
#pragma unroll
  for (int i = 0; i < 8; ++i) { r[i] = lo[i]; r[i + 8] = hi[i]; }
  return r;
}

// out[r][n] = act( mean[r]·Wl[n] + x[r]·Wr[n] + b[n] ), W row-major [OUTD,K].
// Block = 8 waves (4 row-tiles x 2 col-pairs); wave -> 16x32 output tile.
template <int OUTD, bool DROP, bool OUT_BF16>
__global__ __launch_bounds__(256)
void k_sage_gemm(const __bf16* __restrict__ Am, const __bf16* __restrict__ Ax,
                 const __bf16* __restrict__ Wl, const __bf16* __restrict__ Wr,
                 const float* __restrict__ bias, const float* __restrict__ mask,
                 __bf16* __restrict__ outb, float* __restrict__ outf, int nrows) {
  const int K   = DIM;
  int lane = threadIdx.x & 31;
  int wave = threadIdx.x >> 5;
  int wm = wave & 3;                    // row tile within block
  int wn = wave >> 2;                   // col pair within block
  int l15 = lane & 15;
  int hi  = lane >> 4;

  int rowBase = blockIdx.y * 64 + wm * 16;
  int n0 = blockIdx.x * 64 + wn * 32;
  int n1 = n0 + 16;

  int arow = rowBase + l15;
  if (arow > nrows - 1) arow = nrows - 1;   // clamp: EXEC must stay all-ones

  const __bf16* aM  = Am + (size_t)arow * K + hi * 16;
  const __bf16* aX  = Ax + (size_t)arow * K + hi * 16;
  const __bf16* pL0 = Wl + (size_t)(n0 + l15) * K + hi * 16;
  const __bf16* pL1 = Wl + (size_t)(n1 + l15) * K + hi * 16;
  const __bf16* pR0 = Wr + (size_t)(n0 + l15) * K + hi * 16;
  const __bf16* pR1 = Wr + (size_t)(n1 + l15) * K + hi * 16;

  v8f acc0 = {}; v8f acc1 = {};
#pragma unroll
  for (int k0 = 0; k0 < K; k0 += 32) {
    v16bf am  = load_a16(aM + k0);
    v16bf ax  = load_a16(aX + k0);
    v16bf bl0 = load_b16(pL0 + k0);
    v16bf bl1 = load_b16(pL1 + k0);
    v16bf br0 = load_b16(pR0 + k0);
    v16bf br1 = load_b16(pR1 + k0);
    acc0 = __builtin_amdgcn_wmma_f32_16x16x32_bf16(false, am, false, bl0,
                                                   (short)0, acc0, false, false);
    acc1 = __builtin_amdgcn_wmma_f32_16x16x32_bf16(false, am, false, bl1,
                                                   (short)0, acc1, false, false);
    acc0 = __builtin_amdgcn_wmma_f32_16x16x32_bf16(false, ax, false, br0,
                                                   (short)0, acc0, false, false);
    acc1 = __builtin_amdgcn_wmma_f32_16x16x32_bf16(false, ax, false, br1,
                                                   (short)0, acc1, false, false);
  }

  float b0 = bias[n0 + l15];
  float b1 = bias[n1 + l15];
  // C/D layout: VGPR i, lane<16 -> (M=i, N=lane); lane>=16 -> (M=i+8, N=lane-16)
#pragma unroll
  for (int i = 0; i < 8; ++i) {
    int row = rowBase + hi * 8 + i;
    if (row >= nrows) continue;
    float v0 = tanhf(acc0[i] + b0);
    float v1 = tanhf(acc1[i] + b1);
    if (DROP) {
      v0 = (mask[(size_t)row * OUTD + n0 + l15] > 0.2f) ? v0 * 1.25f : 0.0f;
      v1 = (mask[(size_t)row * OUTD + n1 + l15] > 0.2f) ? v1 * 1.25f : 0.0f;
    }
    if (OUT_BF16) {
      outb[(size_t)row * OUTD + n0 + l15] = (__bf16)v0;
      outb[(size_t)row * OUTD + n1 + l15] = (__bf16)v1;
    } else {
      outf[(size_t)row * OUTD + n0 + l15] = v0;
      outf[(size_t)row * OUTD + n1 + l15] = v1;
    }
  }
}

// ------------------------------- driver ------------------------------------

extern "C" void kernel_launch(void* const* d_in, const int* in_sizes, int n_in,
                              void* d_out, int out_size, void* d_ws, size_t ws_size,
                              hipStream_t stream) {
  const float* x     = (const float*)d_in[0];
  const int*   ei    = (const int*)d_in[1];
  const float* mask1 = (const float*)d_in[2];
  const float* mask2 = (const float*)d_in[3];
  const float* W1l   = (const float*)d_in[4];
  const float* b1    = (const float*)d_in[5];
  const float* W1r   = (const float*)d_in[6];
  const float* W2l   = (const float*)d_in[7];
  const float* b2    = (const float*)d_in[8];
  const float* W2r   = (const float*)d_in[9];
  const float* W3l   = (const float*)d_in[10];
  const float* b3    = (const float*)d_in[11];
  const float* W3r   = (const float*)d_in[12];

  const int N = in_sizes[0] / DIM;   // 50000
  const int E = in_sizes[1] / 2;     // 500000
  const int* src = ei;               // edge_index[0]
  const int* dst = ei + E;           // edge_index[1]

  // ---- workspace carve-out (256B aligned) ----
  char* ws = (char*)d_ws;
  size_t off = 0;
  auto carve = [&](size_t bytes) -> void* {
    off = (off + 255) & ~(size_t)255;
    void* p = ws + off;
    off += bytes;
    return p;
  };
  float*  sum   = (float*) carve((size_t)N * DIM * sizeof(float));
  float*  deg   = (float*) carve((size_t)N * sizeof(float));
  __bf16* meanb = (__bf16*)carve((size_t)N * DIM * sizeof(__bf16));
  __bf16* actA  = (__bf16*)carve((size_t)N * DIM * sizeof(__bf16));
  __bf16* actB  = (__bf16*)carve((size_t)N * DIM * sizeof(__bf16));
  __bf16* w1lb  = (__bf16*)carve((size_t)DIM * DIM * sizeof(__bf16));
  __bf16* w1rb  = (__bf16*)carve((size_t)DIM * DIM * sizeof(__bf16));
  __bf16* w2lb  = (__bf16*)carve((size_t)DIM * DIM * sizeof(__bf16));
  __bf16* w2rb  = (__bf16*)carve((size_t)DIM * DIM * sizeof(__bf16));
  __bf16* w3lb  = (__bf16*)carve((size_t)OUTD3 * DIM * sizeof(__bf16));
  __bf16* w3rb  = (__bf16*)carve((size_t)OUTD3 * DIM * sizeof(__bf16));
  (void)ws_size; (void)n_in; (void)out_size;

  const size_t ND = (size_t)N * DIM;

  // ---- degrees (shared by all layers) ----
  k_zero_f32<<<256, 256, 0, stream>>>(deg, (size_t)N);
  k_deg<<<(E + 255) / 256, 256, 0, stream>>>(dst, deg, E);
  k_deg_inv<<<(N + 255) / 256, 256, 0, stream>>>(deg, N);

  // ---- f32 -> bf16 conversions ----
  k_f32_to_bf16<<<4096, 256, 0, stream>>>(x, actA, ND);
  k_f32_to_bf16<<<256, 256, 0, stream>>>(W1l, w1lb, (size_t)DIM * DIM);
  k_f32_to_bf16<<<256, 256, 0, stream>>>(W1r, w1rb, (size_t)DIM * DIM);
  k_f32_to_bf16<<<256, 256, 0, stream>>>(W2l, w2lb, (size_t)DIM * DIM);
  k_f32_to_bf16<<<256, 256, 0, stream>>>(W2r, w2rb, (size_t)DIM * DIM);
  k_f32_to_bf16<<<256, 256, 0, stream>>>(W3l, w3lb, (size_t)OUTD3 * DIM);
  k_f32_to_bf16<<<256, 256, 0, stream>>>(W3r, w3rb, (size_t)OUTD3 * DIM);

  dim3 gemmBlk(256);
  dim3 gemmGrid256(DIM / 64, (N + 63) / 64);     // 4 x 782
  dim3 gemmGrid128(OUTD3 / 64, (N + 63) / 64);   // 2 x 782
  int scatterBlocks = (E + EPB - 1) / EPB;

  // ---- layer 1: x(actA) -> h1(actB) ----
  k_zero_f32<<<4096, 256, 0, stream>>>(sum, ND);
  k_scatter<<<scatterBlocks, DIM, 0, stream>>>(actA, src, dst, sum, E);
  k_mean_bf16<<<4096, 256, 0, stream>>>(sum, deg, meanb, ND);
  k_sage_gemm<DIM, true, true><<<gemmGrid256, gemmBlk, 0, stream>>>(
      meanb, actA, w1lb, w1rb, b1, mask1, actB, nullptr, N);

  // ---- layer 2: h1(actB) -> h2(actA) ----
  k_zero_f32<<<4096, 256, 0, stream>>>(sum, ND);
  k_scatter<<<scatterBlocks, DIM, 0, stream>>>(actB, src, dst, sum, E);
  k_mean_bf16<<<4096, 256, 0, stream>>>(sum, deg, meanb, ND);
  k_sage_gemm<DIM, true, true><<<gemmGrid256, gemmBlk, 0, stream>>>(
      meanb, actB, w2lb, w2rb, b2, mask2, actA, nullptr, N);

  // ---- layer 3: h2(actA) -> d_out (f32, OUT_DIM=128, no dropout) ----
  k_zero_f32<<<4096, 256, 0, stream>>>(sum, ND);
  k_scatter<<<scatterBlocks, DIM, 0, stream>>>(actA, src, dst, sum, E);
  k_mean_bf16<<<4096, 256, 0, stream>>>(sum, deg, meanb, ND);
  k_sage_gemm<OUTD3, false, false><<<gemmGrid128, gemmBlk, 0, stream>>>(
      meanb, actA, w3lb, w3rb, b3, nullptr, nullptr, (float*)d_out, N);
}
